// NodeAttention_66649302499914
// MI455X (gfx1250) — compile-verified, matrix operand order checked by reading, not run on hardware
//
#include <hip/hip_runtime.h>
#include <hip/hip_bf16.h>

typedef __bf16 bf16_t;
typedef __attribute__((ext_vector_type(16))) __bf16 v16bf;
typedef __attribute__((ext_vector_type(8)))  __bf16 v8bf;
typedef __attribute__((ext_vector_type(8)))  float  v8f;

#define NEG_SLOPE 0.2f

// K-pair pattern of the ISA 16-bit A/B fragment layout.
__device__ __forceinline__ int kpair(int p, int half) {
    return ((p < 4) ? (2 * p) : (2 * (p - 4) + 16)) + half * 8;
}

// ---------------- fp32 -> bf16 conversion (node features) ----------------
__global__ void cvt_f32_bf16(const float* __restrict__ src, bf16_t* __restrict__ dst, int n) {
    int i = blockIdx.x * blockDim.x + threadIdx.x;
    if (i < n) dst[i] = (bf16_t)src[i];
}

// ---------------- pack W (f32, [K,Ncol]) into bf16 fragment-major layout ----------------
// Wp[((kc*(Ncol/16) + ct)*32 + lane)*16 + e]  ==  lane `lane`'s e-th bf16 of the B fragment
// for k-chunk kc, col-tile ct.  One thread per (fragment, lane).
__global__ void pack_W(const float* __restrict__ W, bf16_t* __restrict__ Wp, int K, int Ncol) {
    int t = blockIdx.x * blockDim.x + threadIdx.x;
    int nct = Ncol / 16;
    int total = (K / 32) * nct * 32;
    if (t >= total) return;
    int lane = t & 31;
    int frag = t >> 5;
    int ct = frag % nct;
    int kc = frag / nct;
    int half = lane >> 4, lsub = lane & 15;
    int col = ct * 16 + lsub;
    bf16_t* dst = Wp + (size_t)t * 16;
    #pragma unroll
    for (int p = 0; p < 8; ++p) {
        int kb = kpair(p, half) + kc * 32;
        dst[2 * p]     = (bf16_t)W[(size_t)kb * Ncol + col];
        dst[2 * p + 1] = (bf16_t)W[(size_t)(kb + 1) * Ncol + col];
    }
}

// ---------------- WMMA bf16 GEMM:  Y[M,NCOL] = X[M,K] * W[K,NCOL] + bias ----------------
// 4 waves / block, one 16x16 output tile per wave.  W pre-swizzled by pack_W.
template <int K, int NCOL>
__global__ __launch_bounds__(128)
void gemm_wmma_bf16(const bf16_t* __restrict__ X, const bf16_t* __restrict__ Wp,
                    const float* __restrict__ bias, float* __restrict__ Y, int M) {
    const int lane = threadIdx.x & 31;
    const int wave = threadIdx.x >> 5;
    const int half = lane >> 4;
    const int lsub = lane & 15;
    const int rowBase = (blockIdx.x * 4 + wave) * 16;
    if (rowBase >= M) return;                       // wave-uniform: EXEC stays all-ones
    const int colTile = blockIdx.y;
    const int col = colTile * 16 + lsub;

    int arow = rowBase + lsub;
    if (arow >= M) arow = M - 1;                    // clamp ragged loads (stores guarded)

    v8f acc = {};
    #pragma unroll
    for (int kc = 0; kc < K / 32; ++kc) {
        // A fragment: lane's 16 values are bytes [h*16, h*16+16) and [h*16+32, h*16+48)
        // of the row's 64-byte K-window -> two b128 loads.
        const v8bf* Ap = (const v8bf*)(X + (size_t)arow * K + kc * 32 + half * 8);
        union { v16bf v; v8bf h[2]; } A;
        A.h[0] = Ap[0];
        A.h[1] = Ap[2];
        // B fragment: 32 contiguous bytes per lane from the packed buffer.
        v16bf b = *(const v16bf*)(Wp + (((size_t)kc * (NCOL / 16) + colTile) * 32 + lane) * 16);
        acc = __builtin_amdgcn_wmma_f32_16x16x32_bf16(
            false, A.v, false, b, (short)0, acc, false, false);
    }

    const float bcol = bias[col];
    float* yp = Y + (size_t)(rowBase + 8 * half) * NCOL + col;
    if (rowBase + 16 <= M) {                        // fast path: all but the last row-tile
        #pragma unroll
        for (int r = 0; r < 8; ++r) yp[(size_t)r * NCOL] = acc[r] + bcol;
    } else {
        #pragma unroll
        for (int r = 0; r < 8; ++r)
            if (rowBase + 8 * half + r < M) yp[(size_t)r * NCOL] = acc[r] + bcol;
    }
}

// Generic-shape fallback (same packed-W layout, runtime K/Ncol).
__global__ __launch_bounds__(128)
void gemm_wmma_bf16_gen(const bf16_t* __restrict__ X, const bf16_t* __restrict__ Wp,
                        const float* __restrict__ bias, float* __restrict__ Y,
                        int M, int K, int Ncol) {
    const int lane = threadIdx.x & 31;
    const int wave = threadIdx.x >> 5;
    const int half = lane >> 4;
    const int lsub = lane & 15;
    const int rowBase = (blockIdx.x * 4 + wave) * 16;
    if (rowBase >= M) return;
    const int colTile = blockIdx.y;
    const int col = colTile * 16 + lsub;
    int arow = rowBase + lsub;
    if (arow >= M) arow = M - 1;

    v8f acc = {};
    for (int kc = 0; kc < K / 32; ++kc) {
        const v8bf* Ap = (const v8bf*)(X + (size_t)arow * K + kc * 32 + half * 8);
        union { v16bf v; v8bf h[2]; } A;
        A.h[0] = Ap[0];
        A.h[1] = Ap[2];
        v16bf b = *(const v16bf*)(Wp + (((size_t)kc * (Ncol / 16) + colTile) * 32 + lane) * 16);
        acc = __builtin_amdgcn_wmma_f32_16x16x32_bf16(
            false, A.v, false, b, (short)0, acc, false, false);
    }
    const float bcol = bias[col];
    float* yp = Y + (size_t)(rowBase + 8 * half) * Ncol + col;
    #pragma unroll
    for (int r = 0; r < 8; ++r)
        if (rowBase + 8 * half + r < M) yp[(size_t)r * Ncol] = acc[r] + bcol;
}

// ---------------- ordered-uint encoding for float atomic max ----------------
__device__ __forceinline__ unsigned enc_f32(float f) {
    unsigned u = __float_as_uint(f);
    return (u & 0x80000000u) ? ~u : (u | 0x80000000u);
}
__device__ __forceinline__ float dec_f32(unsigned u) {
    u = (u & 0x80000000u) ? (u ^ 0x80000000u) : ~u;
    return __uint_as_float(u);
}

__device__ __forceinline__ float lrelu(float v) { return v > 0.f ? v : NEG_SLOPE * v; }

// ---------------- edge pass 1: logits + segment max ----------------
__global__ void edge_logits_max(const float* __restrict__ xl, const float* __restrict__ xr,
                                const float* __restrict__ att,
                                const long long* __restrict__ srcIdx,
                                const long long* __restrict__ dstIdx,
                                long long E, long long Etot, int H, int C,
                                float* __restrict__ logits, unsigned* __restrict__ mEnc) {
    long long tid = (long long)blockIdx.x * blockDim.x + threadIdx.x;
    if (tid >= Etot * H) return;
    long long e = tid / H; int h = (int)(tid % H);
    long long s = (e < E) ? srcIdx[e] : (e - E);   // self-loops appended after real edges
    long long d = (e < E) ? dstIdx[e] : (e - E);
    const float4* pl = (const float4*)(xl + ((size_t)s * H + h) * C);
    const float4* pr = (const float4*)(xr + ((size_t)d * H + h) * C);
    const float4* pa = (const float4*)(att + (size_t)h * C);
    float acc = 0.f;
    for (int i = 0; i < C / 4; ++i) {
        float4 a = pl[i], b = pr[i], w = pa[i];
        acc += w.x * lrelu(a.x + b.x) + w.y * lrelu(a.y + b.y)
             + w.z * lrelu(a.z + b.z) + w.w * lrelu(a.w + b.w);
    }
    logits[tid] = acc;
    atomicMax(&mEnc[(size_t)d * H + h], enc_f32(acc));
}

// ---------------- edge pass 2: p = exp(logit - max), segment sum ----------------
__global__ void edge_softmax(const long long* __restrict__ dstIdx,
                             long long E, long long Etot, int H,
                             const unsigned* __restrict__ mEnc,
                             float* __restrict__ logits, float* __restrict__ denom) {
    long long tid = (long long)blockIdx.x * blockDim.x + threadIdx.x;
    if (tid >= Etot * H) return;
    long long e = tid / H; int h = (int)(tid % H);
    long long d = (e < E) ? dstIdx[e] : (e - E);
    float mx = dec_f32(mEnc[(size_t)d * H + h]);
    float p = __expf(logits[tid] - mx);
    logits[tid] = p;                                  // overwrite logits with p in place
    atomicAdd(&denom[(size_t)d * H + h], p);
}

// ---------------- edge pass 3: weighted scatter-add of xl[src] ----------------
__global__ void edge_aggregate(const long long* __restrict__ srcIdx,
                               const long long* __restrict__ dstIdx,
                               long long E, long long Etot, int H, int C,
                               const float* __restrict__ p, const float* __restrict__ denom,
                               const float* __restrict__ xl, float* __restrict__ accum) {
    long long tid = (long long)blockIdx.x * blockDim.x + threadIdx.x;
    if (tid >= Etot * H) return;
    long long e = tid / H; int h = (int)(tid % H);
    long long s = (e < E) ? srcIdx[e] : (e - E);
    long long d = (e < E) ? dstIdx[e] : (e - E);
    float alpha = p[tid] / (denom[(size_t)d * H + h] + 1e-16f);
    const float4* xs = (const float4*)(xl + ((size_t)s * H + h) * C);
    float* ac = accum + ((size_t)d * H + h) * C;
    for (int i = 0; i < C / 4; ++i) {
        float4 v = xs[i];
        atomicAdd(&ac[4 * i + 0], v.x * alpha);
        atomicAdd(&ac[4 * i + 1], v.y * alpha);
        atomicAdd(&ac[4 * i + 2], v.z * alpha);
        atomicAdd(&ac[4 * i + 3], v.w * alpha);
    }
}

// ---------------- finalize: head mean + bias + ReLU ----------------
__global__ void finalize_nodes(const float* __restrict__ accum, const float* __restrict__ bias,
                               int N, int H, int C,
                               float* __restrict__ outF, bf16_t* __restrict__ outB) {
    int tid = blockIdx.x * blockDim.x + threadIdx.x;
    if (tid >= N * C) return;
    int n = tid / C, c = tid % C;
    float s = 0.f;
    for (int h = 0; h < H; ++h) s += accum[((size_t)n * H + h) * C + c];
    float v = s / (float)H + bias[c];
    v = v > 0.f ? v : 0.f;                            // reference applies ReLU after each layer
    if (outF) outF[tid] = v;
    if (outB) outB[tid] = (bf16_t)v;
}

// =====================================================================================
extern "C" void kernel_launch(void* const* d_in, const int* in_sizes, int n_in,
                              void* d_out, int out_size, void* d_ws, size_t ws_size,
                              hipStream_t stream) {
    const float*     x    = (const float*)d_in[0];
    const long long* eidx = (const long long*)d_in[1];     // reference dtype: int64
    const float* Wl1 = (const float*)d_in[2],  *bl1 = (const float*)d_in[3];
    const float* Wr1 = (const float*)d_in[4],  *br1 = (const float*)d_in[5];
    const float* att1= (const float*)d_in[6],  *bias1=(const float*)d_in[7];
    const float* Wl2 = (const float*)d_in[8],  *bl2 = (const float*)d_in[9];
    const float* Wr2 = (const float*)d_in[10], *br2 = (const float*)d_in[11];
    const float* att2= (const float*)d_in[12], *bias2=(const float*)d_in[13];

    const int C  = in_sizes[7];                 // 32  (bias1 length)
    const int H  = in_sizes[6] / C;             // 6   (att1 = H*C)
    const int HC = H * C;                       // 192
    const int F  = in_sizes[2] / HC;            // 128 (Wl1 = F*HC)
    const long long N    = (long long)in_sizes[0] / F;   // 50000
    const long long E    = (long long)in_sizes[1] / 2;   // 800000
    const long long Etot = E + N;                        // + self loops
    const long long* srcIdx = eidx;
    const long long* dstIdx = eidx + E;

    // ---- carve workspace (~155 MB) ----
    char* ws = (char*)d_ws;
    size_t off = 0;
    auto carve = [&](size_t bytes) -> char* {
        char* p = ws + off;
        off = (off + bytes + 255) & ~(size_t)255;
        return p;
    };
    bf16_t*   xbf  = (bf16_t*)  carve((size_t)N * F  * sizeof(bf16_t));  // reused as layer-2 input (N*C)
    bf16_t*   Wlp  = (bf16_t*)  carve((size_t)F * HC * sizeof(bf16_t));  // packed fragment-major
    bf16_t*   Wrp  = (bf16_t*)  carve((size_t)F * HC * sizeof(bf16_t));
    float*    xl   = (float*)   carve((size_t)N * HC * sizeof(float));
    float*    xr   = (float*)   carve((size_t)N * HC * sizeof(float));
    float*    lg   = (float*)   carve((size_t)Etot * H * sizeof(float));
    unsigned* mE   = (unsigned*)carve((size_t)N * H * sizeof(unsigned));
    float*    den  = (float*)   carve((size_t)N * H * sizeof(float));
    float*    acc  = (float*)   carve((size_t)N * H * C * sizeof(float));
    (void)ws_size; (void)n_in; (void)out_size;

    const int T = 256;
    auto cd = [](long long n, int b) { return (unsigned)((n + b - 1) / b); };

    const dim3 gemmGrid(cd(N, 64), HC / 16);    // 4 waves/block, 1 tile/wave
    const int  gemmBlk = 128;

    // ================= layer 1 =================
    cvt_f32_bf16<<<cd(N * F, T), T, 0, stream>>>(x, xbf, (int)(N * F));
    pack_W<<<cd((long long)(F / 32) * (HC / 16) * 32, T), T, 0, stream>>>(Wl1, Wlp, F, HC);
    pack_W<<<cd((long long)(F / 32) * (HC / 16) * 32, T), T, 0, stream>>>(Wr1, Wrp, F, HC);
    if (F == 128 && HC == 192) {
        gemm_wmma_bf16<128, 192><<<gemmGrid, gemmBlk, 0, stream>>>(xbf, Wlp, bl1, xl, (int)N);
        gemm_wmma_bf16<128, 192><<<gemmGrid, gemmBlk, 0, stream>>>(xbf, Wrp, br1, xr, (int)N);
    } else {
        gemm_wmma_bf16_gen<<<gemmGrid, gemmBlk, 0, stream>>>(xbf, Wlp, bl1, xl, (int)N, F, HC);
        gemm_wmma_bf16_gen<<<gemmGrid, gemmBlk, 0, stream>>>(xbf, Wrp, br1, xr, (int)N, F, HC);
    }
    hipMemsetAsync(mE,  0, (size_t)N * H * sizeof(unsigned), stream);
    hipMemsetAsync(den, 0, (size_t)N * H * sizeof(float), stream);
    hipMemsetAsync(acc, 0, (size_t)N * H * C * sizeof(float), stream);
    edge_logits_max<<<cd(Etot * H, T), T, 0, stream>>>(xl, xr, att1, srcIdx, dstIdx,
                                                       E, Etot, H, C, lg, mE);
    edge_softmax  <<<cd(Etot * H, T), T, 0, stream>>>(dstIdx, E, Etot, H, mE, lg, den);
    edge_aggregate<<<cd(Etot * H, T), T, 0, stream>>>(srcIdx, dstIdx, E, Etot, H, C,
                                                      lg, den, xl, acc);
    // layer-1 output only needed in bf16 (feeds layer-2 WMMA GEMM); reuse xbf buffer.
    finalize_nodes<<<cd(N * C, T), T, 0, stream>>>(acc, bias1, (int)N, H, C, nullptr, xbf);

    // ================= layer 2 =================
    pack_W<<<cd((long long)(C / 32) * (HC / 16) * 32, T), T, 0, stream>>>(Wl2, Wlp, C, HC);
    pack_W<<<cd((long long)(C / 32) * (HC / 16) * 32, T), T, 0, stream>>>(Wr2, Wrp, C, HC);
    if (C == 32 && HC == 192) {
        gemm_wmma_bf16<32, 192><<<gemmGrid, gemmBlk, 0, stream>>>(xbf, Wlp, bl2, xl, (int)N);
        gemm_wmma_bf16<32, 192><<<gemmGrid, gemmBlk, 0, stream>>>(xbf, Wrp, br2, xr, (int)N);
    } else {
        gemm_wmma_bf16_gen<<<gemmGrid, gemmBlk, 0, stream>>>(xbf, Wlp, bl2, xl, (int)N, C, HC);
        gemm_wmma_bf16_gen<<<gemmGrid, gemmBlk, 0, stream>>>(xbf, Wrp, br2, xr, (int)N, C, HC);
    }
    hipMemsetAsync(mE,  0, (size_t)N * H * sizeof(unsigned), stream);
    hipMemsetAsync(den, 0, (size_t)N * H * sizeof(float), stream);
    hipMemsetAsync(acc, 0, (size_t)N * H * C * sizeof(float), stream);
    edge_logits_max<<<cd(Etot * H, T), T, 0, stream>>>(xl, xr, att2, srcIdx, dstIdx,
                                                       E, Etot, H, C, lg, mE);
    edge_softmax  <<<cd(Etot * H, T), T, 0, stream>>>(dstIdx, E, Etot, H, mE, lg, den);
    edge_aggregate<<<cd(Etot * H, T), T, 0, stream>>>(srcIdx, dstIdx, E, Etot, H, C,
                                                      lg, den, xl, acc);
    finalize_nodes<<<cd(N * C, T), T, 0, stream>>>(acc, bias2, (int)N, H, C,
                                                   (float*)d_out, nullptr);
}
